// SimpleMambaBlock_55963423866972
// MI455X (gfx1250) — matrix operand order repaired and checked by reading, not compiled, not run on hardware
//
#include <hip/hip_runtime.h>
#include <hip/hip_bf16.h>

// ---------------------------------------------------------------------------
// SimpleMambaBlock on MI455X (gfx1250, wave32, WMMA)
//
// Dominant cost is hs(T, E*S) @ Wc.T : 68.7 GFLOP -> compute-bound at bf16
// WMMA rates (bf16 dense >> f32 matrix rate), memory-trivial vs 23.3 TB/s.
// Strategy: bf16 operands, f32 WMMA accumulation (v_wmma_f32_16x16x32_bf16);
// the diagonal scan is parallelized over the 16384 independent (e,s) chains
// and emits hs directly in bf16 (its only consumer is the WMMA GEMM).
//
// GEMM tile traffic uses the gfx1250 async global->LDS path
// (global_load_async_to_lds_b128, ASYNCcnt) with double-buffered LDS so the
// DMA overlaps the WMMA stream; falls back to the proven load+ds_store path
// if the builtin is unavailable on this toolchain.
// ---------------------------------------------------------------------------

typedef __bf16 bf16_t;
typedef __attribute__((ext_vector_type(16))) __bf16 v16bf;
typedef __attribute__((ext_vector_type(8)))  float  v8f;
// exact pointee type expected by the async-LDS builtins (from clang's diag):
typedef __attribute__((__vector_size__(4 * sizeof(int)))) int v4i_t;

#define T_LEN 2048
#define DIM   512
#define E_DIM 1024
#define S_DIM 16
#define KC    4
#define TWO_E 2048
#define ES    (E_DIM * S_DIM)   // 16384

#if defined(__HIP_DEVICE_COMPILE__) && __has_builtin(__builtin_amdgcn_global_load_async_to_lds_b128)
#define USE_ASYNC_LDS 1
#else
#define USE_ASYNC_LDS 0
#endif

#if defined(__HIP_DEVICE_COMPILE__) && __has_builtin(__builtin_amdgcn_s_wait_asynccnt)
#define WAIT_ASYNC(n) __builtin_amdgcn_s_wait_asynccnt(n)
#elif defined(__HIP_DEVICE_COMPILE__)
#define WAIT_ASYNC(n) asm volatile("s_wait_asynccnt %0" ::"i"(n) : "memory")
#else
#define WAIT_ASYNC(n) ((void)0)
#endif

__device__ __forceinline__ float silu_f(float x) {
  return x / (1.0f + __expf(-x));
}

// ---------------------------------------------------------------------------
// f32 -> bf16 conversion
// ---------------------------------------------------------------------------
__global__ void cvt_f32_to_bf16(const float* __restrict__ src,
                                bf16_t* __restrict__ dst, int n) {
  int i = blockIdx.x * blockDim.x + threadIdx.x;
  if (i < n) dst[i] = (bf16_t)src[i];
}

// ---------------------------------------------------------------------------
// Tiled WMMA GEMM: C(M,N) = A(M,K) * B(N,K)^T + bias(N)
// A, B bf16 row-major; C f32. M%128==0, N%128==0, K%32==0 (true for all uses).
// Block 256 threads = 8 waves; macro-tile 128x128, K-step 32.
// Wave (wm 0..3, wn 0..1) owns a 32x64 sub-tile: 2 A-frags x 4 B-frags.
//
// Fragment layouts per CDNA5 ISA 7.12.2 (wave32):
//  A 16x32 bf16: lane L: row = L%16, half = L/16; elements 0..7 hold
//    K = half*8 .. half*8+7 ; elements 8..15 hold K = 16+half*8 .. +7
//    -> two contiguous 16B runs from a row-major [M][K] LDS tile.
//  B 32x16 bf16: lane L: col = L%16, half = L/16; same K mapping as A
//    -> two contiguous 16B runs from row-major [N][K] tile.
//  C/D 16x16 f32: elem r: row = 8*half + r, col = L%16.
// ---------------------------------------------------------------------------
#define GBM 128
#define GBN 128
#define GBK 32
#define GBKP 40   // padded LDS row: 80 bytes (16B-aligned, bank-spread)

__global__ __launch_bounds__(256)
void gemm_bf16_wmma(const bf16_t* __restrict__ A, const bf16_t* __restrict__ B,
                    const float* __restrict__ bias, float* __restrict__ C,
                    int M, int N, int K) {
  const int tid  = threadIdx.x;
  const int wave = tid >> 5;
  const int lane = tid & 31;
  const int wm   = wave & 3;     // M sub-tile: wm*32
  const int wn   = wave >> 2;    // N sub-tile: wn*64
  const int half = lane >> 4;
  const int l16  = lane & 15;

  const int blockN0 = blockIdx.x * GBN;
  const int blockM0 = blockIdx.y * GBM;

  v8f acc[2][4];
  #pragma unroll
  for (int a = 0; a < 2; ++a)
    #pragma unroll
    for (int b = 0; b < 4; ++b)
      #pragma unroll
      for (int r = 0; r < 8; ++r) acc[a][b][r] = 0.0f;

  union Frag { v16bf v; uint4 u[2]; };

#if USE_ASYNC_LDS
  // ---- async global->LDS (ASYNCcnt) with double-buffered tiles ----
  __shared__ __align__(16) bf16_t As[2][GBM][GBKP];
  __shared__ __align__(16) bf16_t Bs[2][GBN][GBKP];

  const int lrow = tid >> 2;        // 0..63  (row pair base, 2 rows/thread)
  const int lseg = tid & 3;         // 16B segment within 64B row

  // issue one 128x32 A tile + 128x32 B tile into buffer `bf` (4 b128/thread)
  auto issue_tile = [&](int bf, int kk) {
    #pragma unroll
    for (int i = 0; i < 2; ++i) {
      int row = lrow + i * 64;
      __builtin_amdgcn_global_load_async_to_lds_b128(
          (__attribute__((address_space(1))) v4i_t*)(
              A + (size_t)(blockM0 + row) * K + kk + lseg * 8),
          (__attribute__((address_space(3))) v4i_t*)&As[bf][row][lseg * 8],
          0, 0);
      __builtin_amdgcn_global_load_async_to_lds_b128(
          (__attribute__((address_space(1))) v4i_t*)(
              B + (size_t)(blockN0 + row) * K + kk + lseg * 8),
          (__attribute__((address_space(3))) v4i_t*)&Bs[bf][row][lseg * 8],
          0, 0);
    }
  };

  const int nsteps = K / GBK;
  issue_tile(0, 0);
  for (int it = 0; it < nsteps; ++it) {
    const int cur = it & 1;
    const bool more = (it + 1) < nsteps;
    if (more) issue_tile(cur ^ 1, (it + 1) * GBK);
    // older 4 async ops (current buffer) complete; newer 4 still in flight
    if (more) WAIT_ASYNC(4); else WAIT_ASYNC(0);
    __syncthreads();

    Frag af[2], bfr[4];
    #pragma unroll
    for (int a = 0; a < 2; ++a) {
      int r = wm * 32 + a * 16 + l16;
      af[a].u[0] = *(const uint4*)&As[cur][r][half * 8];
      af[a].u[1] = *(const uint4*)&As[cur][r][16 + half * 8];
    }
    #pragma unroll
    for (int b = 0; b < 4; ++b) {
      int n = wn * 64 + b * 16 + l16;
      bfr[b].u[0] = *(const uint4*)&Bs[cur][n][half * 8];
      bfr[b].u[1] = *(const uint4*)&Bs[cur][n][16 + half * 8];
    }
    #pragma unroll
    for (int a = 0; a < 2; ++a)
      #pragma unroll
      for (int b = 0; b < 4; ++b)
        acc[a][b] = __builtin_amdgcn_wmma_f32_16x16x32_bf16(
            false, af[a].v, false, bfr[b].v, (short)0, acc[a][b], false, false);
    __syncthreads();   // all reads of `cur` done before it is refilled
  }
#else
  // ---- fallback: global->VGPR->ds_store, single buffer ----
  __shared__ __align__(16) bf16_t As[GBM][GBKP];
  __shared__ __align__(16) bf16_t Bs[GBN][GBKP];

  for (int kk = 0; kk < K; kk += GBK) {
    #pragma unroll
    for (int i = 0; i < 2; ++i) {
      int l   = tid + i * 256;
      int row = l >> 2;
      int seg = l & 3;
      *(uint4*)&As[row][seg * 8] =
          *(const uint4*)(A + (size_t)(blockM0 + row) * K + kk + seg * 8);
      *(uint4*)&Bs[row][seg * 8] =
          *(const uint4*)(B + (size_t)(blockN0 + row) * K + kk + seg * 8);
    }
    __syncthreads();

    Frag af[2], bfr[4];
    #pragma unroll
    for (int a = 0; a < 2; ++a) {
      int r = wm * 32 + a * 16 + l16;
      af[a].u[0] = *(const uint4*)&As[r][half * 8];
      af[a].u[1] = *(const uint4*)&As[r][16 + half * 8];
    }
    #pragma unroll
    for (int b = 0; b < 4; ++b) {
      int n = wn * 64 + b * 16 + l16;
      bfr[b].u[0] = *(const uint4*)&Bs[n][half * 8];
      bfr[b].u[1] = *(const uint4*)&Bs[n][16 + half * 8];
    }
    #pragma unroll
    for (int a = 0; a < 2; ++a)
      #pragma unroll
      for (int b = 0; b < 4; ++b)
        acc[a][b] = __builtin_amdgcn_wmma_f32_16x16x32_bf16(
            false, af[a].v, false, bfr[b].v, (short)0, acc[a][b], false, false);
    __syncthreads();
  }
#endif

  // writeback (+bias), C/D layout: elem r -> row 8*half + r, col = l16
  #pragma unroll
  for (int a = 0; a < 2; ++a) {
    int row0 = blockM0 + wm * 32 + a * 16 + half * 8;
    #pragma unroll
    for (int b = 0; b < 4; ++b) {
      int col  = blockN0 + wn * 64 + b * 16 + l16;
      float bv = bias ? bias[col] : 0.0f;
      #pragma unroll
      for (int r = 0; r < 8; ++r)
        C[(size_t)(row0 + r) * N + col] = acc[a][b][r] + bv;
    }
  }
}

// ---------------------------------------------------------------------------
// Depthwise SAME conv (K=4, pad_lo=1) over time + SiLU -> u; silu(gate) -> gate
// xp is (T, 2E): gate half = xp[:, :E], proj half = xp[:, E:2E]
// ---------------------------------------------------------------------------
__global__ void conv_silu_gate(const float* __restrict__ xp,
                               const float* __restrict__ conv_w,
                               const float* __restrict__ conv_b,
                               float* __restrict__ u,
                               float* __restrict__ gate) {
  int idx = blockIdx.x * blockDim.x + threadIdx.x;  // t*E + e
  int t = idx / E_DIM, e = idx % E_DIM;
  float accv = conv_b[e];
  #pragma unroll
  for (int k = 0; k < KC; ++k) {
    int tt = t - 1 + k;
    if (tt >= 0 && tt < T_LEN)
      accv += conv_w[e * KC + k] * xp[(size_t)tt * TWO_E + E_DIM + e];
  }
  u[idx]    = silu_f(accv);
  gate[idx] = silu_f(xp[(size_t)t * TWO_E + e]);
}

// ---------------------------------------------------------------------------
// B_const[i] = sum_j Wb[i, j] + bb[i], i in [0, E*S). One wave32 per row.
// ---------------------------------------------------------------------------
__global__ void bconst_rowsum(const float* __restrict__ Wb,
                              const float* __restrict__ bb,
                              float* __restrict__ Bconst) {
  int wave = threadIdx.x >> 5, lane = threadIdx.x & 31;
  int row = blockIdx.x * 8 + wave;
  float s = 0.0f;
  for (int j = lane; j < E_DIM; j += 32) s += Wb[(size_t)row * E_DIM + j];
  #pragma unroll
  for (int off = 16; off > 0; off >>= 1) s += __shfl_xor(s, off, 32);
  if (lane == 0) Bconst[row] = s + bb[row];
}

// ---------------------------------------------------------------------------
// Diagonal linear recurrence, parallel over the 16384 independent (e,s)
// chains; emits hs in bf16 (sole consumer is the WMMA GEMM).
// ---------------------------------------------------------------------------
__global__ void scan_state(const float* __restrict__ u,
                           const float* __restrict__ A_in,
                           const float* __restrict__ Bconst,
                           bf16_t* __restrict__ hs) {
  int idx = blockIdx.x * blockDim.x + threadIdx.x;  // e*S + s
  int e = idx / S_DIM;
  float decay = __expf(A_in[idx]);
  float Bv = Bconst[idx];
  float h = 0.0f;
  for (int t = 0; t < T_LEN; ++t) {
    h = h * decay + u[(size_t)t * E_DIM + e] * Bv;
    hs[(size_t)t * ES + idx] = (bf16_t)h;
  }
}

// ---------------------------------------------------------------------------
// y = (ygemm(+bc already) + u*D) * silu(gate); emit bf16 for final GEMM
// ---------------------------------------------------------------------------
__global__ void y_gate_cvt(const float* __restrict__ ygemm,
                           const float* __restrict__ u,
                           const float* __restrict__ D,
                           const float* __restrict__ gate,
                           bf16_t* __restrict__ ybf) {
  int idx = blockIdx.x * blockDim.x + threadIdx.x;
  int e = idx % E_DIM;
  ybf[idx] = (bf16_t)((ygemm[idx] + u[idx] * D[e]) * gate[idx]);
}

// ---------------------------------------------------------------------------
extern "C" void kernel_launch(void* const* d_in, const int* in_sizes, int n_in,
                              void* d_out, int out_size, void* d_ws, size_t ws_size,
                              hipStream_t stream) {
  const float* x      = (const float*)d_in[0];
  const float* W_in   = (const float*)d_in[1];
  const float* b_in   = (const float*)d_in[2];
  const float* conv_w = (const float*)d_in[3];
  const float* conv_b = (const float*)d_in[4];
  const float* A      = (const float*)d_in[5];
  const float* Wb     = (const float*)d_in[6];
  const float* bb     = (const float*)d_in[7];
  const float* Wc     = (const float*)d_in[8];
  const float* bc     = (const float*)d_in[9];
  const float* D      = (const float*)d_in[10];
  const float* W_out  = (const float*)d_in[11];
  const float* b_out  = (const float*)d_in[12];
  float* out = (float*)d_out;
  (void)in_sizes; (void)n_in; (void)out_size; (void)ws_size;

  char* wsp = (char*)d_ws;
  auto alloc = [&](size_t bytes) -> char* {
    char* p = wsp;
    wsp += (bytes + 255) & ~(size_t)255;
    return p;
  };
  bf16_t* x_bf    = (bf16_t*)alloc((size_t)T_LEN * DIM * 2);
  bf16_t* Win_bf  = (bf16_t*)alloc((size_t)TWO_E * DIM * 2);
  bf16_t* Wc_bf   = (bf16_t*)alloc((size_t)E_DIM * ES * 2);
  bf16_t* Wout_bf = (bf16_t*)alloc((size_t)DIM * E_DIM * 2);
  float*  xp      = (float*) alloc((size_t)T_LEN * TWO_E * 4);
  float*  u       = (float*) alloc((size_t)T_LEN * E_DIM * 4);
  float*  gate    = (float*) alloc((size_t)T_LEN * E_DIM * 4);
  float*  Bconst  = (float*) alloc((size_t)ES * 4);
  bf16_t* hs_bf   = (bf16_t*)alloc((size_t)T_LEN * ES * 2);
  float*  ygemm   = (float*) alloc((size_t)T_LEN * E_DIM * 4);
  bf16_t* y_bf    = (bf16_t*)alloc((size_t)T_LEN * E_DIM * 2);

  // --- weight/input conversions to bf16 ---
  cvt_f32_to_bf16<<<(T_LEN * DIM) / 256, 256, 0, stream>>>(x, x_bf, T_LEN * DIM);
  cvt_f32_to_bf16<<<(TWO_E * DIM) / 256, 256, 0, stream>>>(W_in, Win_bf, TWO_E * DIM);
  cvt_f32_to_bf16<<<(E_DIM * ES) / 256, 256, 0, stream>>>(Wc, Wc_bf, E_DIM * ES);
  cvt_f32_to_bf16<<<(DIM * E_DIM) / 256, 256, 0, stream>>>(W_out, Wout_bf, DIM * E_DIM);

  // --- GEMM1: xp = x @ W_in^T + b_in  (2048 x 2048 x 512) ---
  gemm_bf16_wmma<<<dim3(TWO_E / GBN, T_LEN / GBM), 256, 0, stream>>>(
      x_bf, Win_bf, b_in, xp, T_LEN, TWO_E, DIM);

  // --- conv + SiLU + gate ---
  conv_silu_gate<<<(T_LEN * E_DIM) / 256, 256, 0, stream>>>(xp, conv_w, conv_b, u, gate);

  // --- B_const = rowsum(Wb) + bb ---
  bconst_rowsum<<<ES / 8, 256, 0, stream>>>(Wb, bb, Bconst);

  // --- diagonal scan -> hs (bf16) ---
  scan_state<<<ES / 256, 256, 0, stream>>>(u, A, Bconst, hs_bf);

  // --- GEMM2 (dominant): ygemm = hs @ Wc^T + bc  (2048 x 1024 x 16384) ---
  gemm_bf16_wmma<<<dim3(E_DIM / GBN, T_LEN / GBM), 256, 0, stream>>>(
      hs_bf, Wc_bf, bc, ygemm, T_LEN, E_DIM, ES);

  // --- elementwise: (+ u*D) * silu(gate), -> bf16 ---
  y_gate_cvt<<<(T_LEN * E_DIM) / 256, 256, 0, stream>>>(ygemm, u, D, gate, y_bf);

  // --- GEMM3: out = y @ W_out^T + b_out  (2048 x 512 x 1024) ---
  gemm_bf16_wmma<<<dim3(DIM / GBN, T_LEN / GBM), 256, 0, stream>>>(
      y_bf, Wout_bf, b_out, out, T_LEN, DIM, E_DIM);
}
